// VisualAttention_6957847019570
// MI455X (gfx1250) — matrix operand order repaired and checked by reading, not compiled, not run on hardware
//
#include <hip/hip_runtime.h>

// ============================================================================
// VisualAttention fused kernel set for MI455X (gfx1250, wave32, WMMA).
//
//  K1: fp32 -> bf16 conversion of text / obj (streaming)
//  K2: W transpose + bf16 (tiny, 768x512)
//  K3: projection GEMM  t = text @ W + b  (v_wmma_f32_16x16x32_bf16) -> ws
//  K3b: LDS-tiled transpose t -> tT[d][l]  (coalesced both sides)
//  K4: fused flash-attention: softmax(obj @ t^T) @ t -> d_out (fp32)
//
//  Rationale (compute-bound: ~95 GFLOP vs ~150 MB HBM): bf16 WMMA gives 8x
//  the K-depth per instruction of fp32 WMMA while keeping fp32 exponent.
//  Per-batch t/tT (8 MB bf16) is L2-resident (192 MB L2), so all GEMM
//  operands stream straight from L2 as contiguous b128 fragments that the
//  compiler clauses + pipelines around the WMMAs — no LDS staging, and no
//  opaque asm loads in the hot loop.
// ============================================================================

typedef __bf16 bf16;
typedef __attribute__((ext_vector_type(16))) __bf16 v16bf;
typedef __attribute__((ext_vector_type(8)))  __bf16 v8bf;
typedef __attribute__((ext_vector_type(4)))  __bf16 v4bf;
typedef __attribute__((ext_vector_type(8)))  float  v8f;
typedef __attribute__((ext_vector_type(4)))  float  v4f;
typedef __attribute__((ext_vector_type(4)))  int    v4i;
typedef __attribute__((ext_vector_type(8)))  int    v8i;

#define DEV static __device__ __forceinline__

// ---- sizes ------------------------------------------------------------------
#define BATCH 8
#define NUM   1024
#define IDF   512
#define CDF   768
#define LTOT  4096          // 4 captions * 1024
#define MTOT  (BATCH*LTOT)  // projection rows
#define CHUNK 64            // L-chunk in the fused attention kernel

// ---- workspace layout (bytes), total ~127 MB --------------------------------
#define WS_TEXTBF 0u                                   // MTOT*CDF  bf16 = 50331648
#define WS_TBF    (50331648u)                          // MTOT*IDF  bf16 = 33554432
#define WS_OBJBF  (50331648u + 33554432u)              // B*NUM*IDF bf16 =  8388608
#define WS_WT     (50331648u + 33554432u + 8388608u)   // IDF*CDF   bf16 =   786432
#define WS_TT     (50331648u + 33554432u + 8388608u + 786432u)  // [B][IDF][LTOT] bf16

// ---- fragment helpers -------------------------------------------------------
DEV v16bf combine_frag(v4i lo, v4i hi) {
  v8i t;
  t[0]=lo[0]; t[1]=lo[1]; t[2]=lo[2]; t[3]=lo[3];
  t[4]=hi[0]; t[5]=hi[1]; t[6]=hi[2]; t[7]=hi[3];
  return __builtin_bit_cast(v16bf, t);
}

// two 16B loads (global or LDS; compiler emits global_load_b128 / ds_load_b128)
DEV v16bf load_frag(const bf16* p0, const bf16* p1) {
  v4i lo = *(const v4i*)p0;
  v4i hi = *(const v4i*)p1;
  return combine_frag(lo, hi);
}

DEV v8f wmma_bf16(v16bf a, v16bf b, v8f c) {
  return __builtin_amdgcn_wmma_f32_16x16x32_bf16(
      /*neg_a=*/false, a, /*neg_b=*/false, b,
      /*c_mod=*/(short)0, c, /*reuse_a=*/false, /*reuse_b=*/false);
}

// ============================================================================
// K1: fp32 -> bf16 elementwise (vector-4)
// ============================================================================
__global__ void va_cvt_bf16(const float* __restrict__ in,
                            bf16* __restrict__ out, int n) {
  int i = (blockIdx.x * blockDim.x + threadIdx.x) * 4;
  if (i < n) {
    v4f v = *(const v4f*)(in + i);
    v4bf o;
    o[0] = (bf16)v[0]; o[1] = (bf16)v[1]; o[2] = (bf16)v[2]; o[3] = (bf16)v[3];
    *(v4bf*)(out + i) = o;
  }
}

// ============================================================================
// K2: WT[n][k] = (bf16) W[k][n]   (W: [CDF][IDF]) — tiny, L2-resident
// ============================================================================
__global__ void va_transpose_w(const float* __restrict__ W,
                               bf16* __restrict__ WT) {
  int idx = blockIdx.x * blockDim.x + threadIdx.x;
  if (idx < IDF * CDF) {
    int n = idx / CDF, k = idx % CDF;
    WT[idx] = (bf16)W[k * IDF + n];
  }
}

// ============================================================================
// K3: projection GEMM  t = textbf(MTOT x CDF) @ W(CDF x IDF) + bias -> tbf
// block 256 = 8 waves; tile 64(M) x 64(N); wave: 16 rows x 32 cols.
// ============================================================================
__global__ void __launch_bounds__(256, 2)
va_proj(const bf16* __restrict__ textbf, const bf16* __restrict__ WT,
        const float* __restrict__ bias, bf16* __restrict__ tbf) {
  const int tid  = threadIdx.x;
  const int w    = tid >> 5, lane = tid & 31;
  const int hi   = lane >> 4, ln = lane & 15;
  const int mbase = blockIdx.x * 64 + (w >> 1) * 16;
  const int nbase = blockIdx.y * 64 + (w & 1) * 32;

  v8f acc0 = {0,0,0,0,0,0,0,0}, acc1 = acc0;

  const bf16* arow = textbf + (size_t)(mbase + ln) * CDF;
  const int   koff = hi * 16;          // A: {0..15} / {16..31}

  #pragma unroll 4
  for (int kb = 0; kb < CDF; kb += 32) {
    v16bf a = load_frag(arow + kb + koff, arow + kb + koff + 8);
    const bf16* b0 = WT + (size_t)(nbase +      ln) * CDF + kb + hi * 16;
    const bf16* b1 = WT + (size_t)(nbase + 16 + ln) * CDF + kb + hi * 16;
    acc0 = wmma_bf16(a, load_frag(b0, b0 + 8), acc0);
    acc1 = wmma_bf16(a, load_frag(b1, b1 + 8), acc1);
  }

  const float bv0 = bias[nbase + ln];
  const float bv1 = bias[nbase + 16 + ln];
  #pragma unroll
  for (int r = 0; r < 8; ++r) {
    int row = mbase + r + hi * 8;     // C layout: VGPR r -> row r+8*hi
    tbf[(size_t)row * IDF + nbase + ln]      = (bf16)(acc0[r] + bv0);
    tbf[(size_t)row * IDF + nbase + 16 + ln] = (bf16)(acc1[r] + bv1);
  }
}

// ============================================================================
// K3b: per-batch transpose tbf [4096][512] -> tT [512][4096], 64x64 LDS tiles.
// Both global sides fully coalesced; LDS pad keeps b128 alignment + banks.
// ============================================================================
__global__ void va_transpose_t(const bf16* __restrict__ tbf,
                               bf16* __restrict__ tT) {
  __shared__ bf16 tile[64][72];       // [d][l], 144B row stride (16B aligned)
  const int tid = threadIdx.x;
  const int l0 = blockIdx.x * 64, d0 = blockIdx.y * 64;
  const bf16* in = tbf + (size_t)blockIdx.z * LTOT * IDF;
  bf16* outp     = tT  + (size_t)blockIdx.z * IDF * LTOT;

  #pragma unroll
  for (int p = 0; p < 2; ++p) {
    const int l = p * 32 + (tid >> 3);        // 32 l-rows per pass
    const int dblk = (tid & 7) * 8;
    v4i v = *(const v4i*)(in + (size_t)(l0 + l) * IDF + d0 + dblk);
    v8bf e = __builtin_bit_cast(v8bf, v);
    #pragma unroll
    for (int i = 0; i < 8; ++i) tile[dblk + i][l] = e[i];
  }
  __syncthreads();
  const int d = tid >> 2;                     // 0..63
  const int lblk = (tid & 3) * 16;
  v4i a = *(const v4i*)&tile[d][lblk];
  v4i b = *(const v4i*)&tile[d][lblk + 8];
  bf16* op = outp + (size_t)(d0 + d) * LTOT + l0 + lblk;
  *(v4i*)op       = a;
  *(v4i*)(op + 8) = b;
}

// ============================================================================
// K4: fused attention. Per block: 64 obj rows of one batch, full d=512.
// 8 waves: (rowgroup 0..3) x (d-half 0..1); wave owns 16 rows x 256 cols.
// Online softmax over L=4096 in chunks of 64; score work duplicated across
// the d-half pair -> zero inter-wave sync; LDS only for the per-wave P tile.
// ============================================================================
__global__ void __launch_bounds__(256, 1)
va_attn(const bf16* __restrict__ objbf, const bf16* __restrict__ tbf,
        const bf16* __restrict__ tT, float* __restrict__ out) {
  const int tid  = threadIdx.x;
  const int w    = tid >> 5, lane = tid & 31;
  const int hi   = lane >> 4, ln = lane & 15;
  const int bi   = blockIdx.y;
  const int rowbase = blockIdx.x * 64 + (w >> 1) * 16;   // row within batch
  const int dbase   = (w & 1) * 256;

  const bf16* objr = objbf + ((size_t)(bi * NUM + rowbase + ln)) * IDF;
  const bf16* tb   = tbf + (size_t)bi * LTOT * IDF;      // [4096][512]
  const bf16* ttb  = tT  + (size_t)bi * IDF * LTOT;      // [512][4096]

  __shared__ bf16 Pl[8][16][CHUNK];                      // per-wave P tile

  v8f accO[16];
  #pragma unroll
  for (int nt = 0; nt < 16; ++nt) accO[nt] = (v8f){0,0,0,0,0,0,0,0};
  float m[8], s[8];
  #pragma unroll
  for (int r = 0; r < 8; ++r) { m[r] = -3.0e38f; s[r] = 0.0f; }

  const int koffA = hi * 8;

  for (int l0 = 0; l0 < LTOT; l0 += CHUNK) {
    // ---- S = obj(16x512) @ t^T(512x64), K swept in 16 steps -------------
    v8f sc[4];
    #pragma unroll
    for (int j = 0; j < 4; ++j) sc[j] = (v8f){0,0,0,0,0,0,0,0};
    #pragma unroll 4
    for (int kb = 0; kb < IDF; kb += 32) {
      v16bf a = load_frag(objr + kb + hi * 16, objr + kb + hi * 16 + 8);
      #pragma unroll
      for (int j = 0; j < 4; ++j) {
        const bf16* bp = tb + (size_t)(l0 + j * 16 + ln) * IDF + kb + hi * 16;
        sc[j] = wmma_bf16(a, load_frag(bp, bp + 8), sc[j]);
      }
    }

    // ---- online softmax row update (rows per-VGPR, cols per-lane) -------
    float scale[8];
    #pragma unroll
    for (int r = 0; r < 8; ++r) {
      float v = fmaxf(fmaxf(sc[0][r], sc[1][r]), fmaxf(sc[2][r], sc[3][r]));
      v = fmaxf(v, __shfl_xor(v, 1, 16));
      v = fmaxf(v, __shfl_xor(v, 2, 16));
      v = fmaxf(v, __shfl_xor(v, 4, 16));
      v = fmaxf(v, __shfl_xor(v, 8, 16));
      const float mn = fmaxf(m[r], v);
      const float scf = __expf(m[r] - mn);
      float rs = 0.0f;
      #pragma unroll
      for (int j = 0; j < 4; ++j) {
        const float p = __expf(sc[j][r] - mn);
        Pl[w][r + hi * 8][j * 16 + ln] = (bf16)p;
        rs += p;
      }
      rs += __shfl_xor(rs, 1, 16);
      rs += __shfl_xor(rs, 2, 16);
      rs += __shfl_xor(rs, 4, 16);
      rs += __shfl_xor(rs, 8, 16);
      s[r] = s[r] * scf + rs;
      m[r] = mn;
      scale[r] = scf;
    }
    #pragma unroll
    for (int nt = 0; nt < 16; ++nt)
      #pragma unroll
      for (int r = 0; r < 8; ++r) accO[nt][r] *= scale[r];

    // same-wave DS ops are in-order; wait + compiler barrier before the
    // cross-lane re-read of P
    asm volatile("s_wait_dscnt 0x0" ::: "memory");

    // ---- O += P(16x64) @ t_chunk(64x256-half), two K steps --------------
    #pragma unroll
    for (int ks = 0; ks < 2; ++ks) {
      const bf16* pr = &Pl[w][ln][0] + ks * 32 + hi * 16;
      v16bf ap = load_frag(pr, pr + 8);
      #pragma unroll
      for (int nt = 0; nt < 16; ++nt) {
        // B[k=l][n=d] from tT[d][l]: contiguous K per lane -> plain b128s
        const bf16* q = ttb + (size_t)(dbase + nt * 16 + ln) * LTOT
                            + l0 + ks * 32 + hi * 16;
        accO[nt] = wmma_bf16(ap, load_frag(q, q + 8), accO[nt]);
      }
    }
  }

  // ---- epilogue: O /= rowsum, fp32 store --------------------------------
  float* obase = out + (size_t)bi * NUM * IDF;
  #pragma unroll
  for (int r = 0; r < 8; ++r) {
    const float inv = 1.0f / s[r];
    const int row = rowbase + r + hi * 8;
    #pragma unroll
    for (int nt = 0; nt < 16; ++nt)
      obase[(size_t)row * IDF + dbase + nt * 16 + ln] = accO[nt][r] * inv;
  }
}

// ============================================================================
extern "C" void kernel_launch(void* const* d_in, const int* in_sizes, int n_in,
                              void* d_out, int out_size, void* d_ws, size_t ws_size,
                              hipStream_t stream) {
  const float* obj  = (const float*)d_in[0];   // [8,1024,512]
  const float* text = (const float*)d_in[1];   // [8,4,1024,768]
  const float* W    = (const float*)d_in[2];   // [768,512]
  const float* bias = (const float*)d_in[3];   // [512]
  float* out = (float*)d_out;                  // [8,1024,512]

  char* ws = (char*)d_ws;
  bf16* textbf = (bf16*)(ws + WS_TEXTBF);
  bf16* tbf    = (bf16*)(ws + WS_TBF);
  bf16* objbf  = (bf16*)(ws + WS_OBJBF);
  bf16* WT     = (bf16*)(ws + WS_WT);
  bf16* tT     = (bf16*)(ws + WS_TT);

  const int n_text = MTOT * CDF;          // 25165824
  const int n_obj  = BATCH * NUM * IDF;   // 4194304

  va_cvt_bf16<<<(n_text/4 + 255)/256, 256, 0, stream>>>(text, textbf, n_text);
  va_cvt_bf16<<<(n_obj /4 + 255)/256, 256, 0, stream>>>(obj,  objbf,  n_obj);
  va_transpose_w<<<(IDF*CDF + 255)/256, 256, 0, stream>>>(W, WT);

  va_proj<<<dim3(MTOT/64, IDF/64), 256, 0, stream>>>(textbf, WT, bias, tbf);
  va_transpose_t<<<dim3(LTOT/64, IDF/64, BATCH), 256, 0, stream>>>(tbf, tT);
  va_attn<<<dim3(NUM/64, BATCH), 256, 0, stream>>>(objbf, tbf, tT, out);
}
